// TemporalMambaFusionNet_89945205113402
// MI455X (gfx1250) — compile-verified, matrix operand order checked by reading, not run on hardware
//
#include <hip/hip_runtime.h>
#include <math.h>

// ---------------------------------------------------------------------------
// CDNA5 (gfx1250) types
// ---------------------------------------------------------------------------
typedef __bf16 bf16_t;
typedef __attribute__((ext_vector_type(16))) __bf16 v16bf;
typedef __attribute__((ext_vector_type(8)))  float  v8f;
typedef __attribute__((ext_vector_type(4)))  float  f32x4;

__device__ __forceinline__ bf16_t f2bf(float f) {
    union { float f; unsigned u; } v; v.f = f;
    unsigned r = v.u + 0x7FFFu + ((v.u >> 16) & 1u);   // round-to-nearest-even
    unsigned short h = (unsigned short)(r >> 16);
    bf16_t out;
    __builtin_memcpy(&out, &h, 2);
    return out;
}

__device__ __forceinline__ float gelu_exact(float x) {
    return 0.5f * x * (1.0f + erff(x * 0.7071067811865476f));
}
__device__ __forceinline__ float sigm(float x) { return 1.0f / (1.0f + expf(-x)); }

#define CEILDIV(a, b) (((a) + (b) - 1) / (b))

// Fragment K-offset map per CDNA5 ISA 7.12.2 (wave32, 16-bit A/B operands):
// lane holds row/col = lane&15; elem e -> K = koff(e, lane>>4)
__device__ __forceinline__ int koff(int e, int half) {
    return (e < 8) ? (half * 8 + e) : (8 + half * 8 + e);   // == 16+half*8+(e-8)
}

// ---------------------------------------------------------------------------
// Implicit-GEMM 2D conv (NCHW, OIHW weights), bf16 WMMA, f32 accumulate.
//
// Reduction order: (kh, kw) outer, cin in chunks of 32 inner -> NO integer
// division in the hot loop (weight gather stride = k*k, activation gather
// stride = Hin*Win, border validity hoisted per (kh,kw)).
//
// All 4 waves of a block share the same Cout tile (m0), so the 16x32 bf16
// weight fragment is staged cooperatively in LDS (double-buffered, one
// barrier per chunk): 128 threads x 4 elements -> ds_store_b64 each, laid
// out so each lane reads its full A fragment as 32 contiguous bytes
// (2 x ds_load_b128). Activations are gathered per lane (coalesced across
// lanes 0-15 per channel). One wave computes a 16(Cout) x 32(spatial) D tile
// = 2 accumulators sharing one A fragment -> 2 v_wmma per K-chunk.
// ---------------------------------------------------------------------------
__global__ __launch_bounds__(128)
void conv_wmma_k(const float* __restrict__ x, const float* __restrict__ w,
                 const float* __restrict__ bias, float* __restrict__ y,
                 int N, int Cin, int Hin, int Win, int Cout, int Hout, int Wout,
                 int stride, int pad, int ksz)
{
    __shared__ alignas(32) bf16_t sha[2][512];     // 2 x (16 rows x 32 K) bf16

    const int tid  = threadIdx.x;
    const int lane = tid & 31;
    const int wv   = tid >> 5;
    const int l16  = lane & 15;
    const int half = lane >> 4;
    const int HWo  = Hout * Wout;
    const int HWi  = Hin * Win;
    const int kk2  = ksz * ksz;

    const int tileS = blockIdx.x * 128 + wv * 32;
    const int m0    = blockIdx.y * 16;
    const int img   = blockIdx.z;

    // two spatial sub-tiles per wave
    const int s0 = tileS + l16;
    const int s1 = s0 + 16;
    const bool sv0 = (s0 < HWo), sv1 = (s1 < HWo);
    const int oh0 = sv0 ? (s0 / Wout) : 0, ow0 = sv0 ? (s0 - oh0 * Wout) : 0;
    const int oh1 = sv1 ? (s1 / Wout) : 0, ow1 = sv1 ? (s1 - oh1 * Wout) : 0;

    const float* xim = x + (size_t)img * Cin * HWi;

    // writer-side decomposition (loop-invariant): thread writes LDS elements
    // 4*tid .. 4*tid+3; element idx -> (lane = idx>>4, e = idx&15) ->
    // weight (row = lane&15, ci = cc + koff(e, lane>>4))
    const float* wrptr[4];
    int  wkof[4];
    bool wval[4];
#pragma unroll
    for (int j = 0; j < 4; ++j) {
        const int idx  = 4 * tid + j;
        const int wl   = idx >> 4;
        const int we   = idx & 15;
        const int wrow = m0 + (wl & 15);
        wkof[j]  = koff(we, (wl >> 4) & 1);
        wval[j]  = (wrow < Cout);
        wrptr[j] = w + (size_t)wrow * Cin * kk2;
    }

    v8f acc0 = {0.f, 0.f, 0.f, 0.f, 0.f, 0.f, 0.f, 0.f};
    v8f acc1 = {0.f, 0.f, 0.f, 0.f, 0.f, 0.f, 0.f, 0.f};

    int bufc = 0;
    for (int kh = 0; kh < ksz; ++kh) {
        for (int kw_ = 0; kw_ < ksz; ++kw_) {
            const int kpos = kh * ksz + kw_;
            // hoisted border validity + flat offsets per sub-tile
            const int ih0 = oh0 * stride - pad + kh;
            const int iw0 = ow0 * stride - pad + kw_;
            const int ih1 = oh1 * stride - pad + kh;
            const int iw1 = ow1 * stride - pad + kw_;
            const bool v0 = sv0 && ih0 >= 0 && ih0 < Hin && iw0 >= 0 && iw0 < Win;
            const bool v1 = sv1 && ih1 >= 0 && ih1 < Hin && iw1 >= 0 && iw1 < Win;
            const int off0 = v0 ? (ih0 * Win + iw0) : 0;
            const int off1 = v1 ? (ih1 * Win + iw1) : 0;

            for (int cc = 0; cc < Cin; cc += 32) {
                // --- cooperative weight stage into LDS (double buffered) ---
                bf16_t tmp4[4];
#pragma unroll
                for (int j = 0; j < 4; ++j) {
                    const int ci = cc + wkof[j];
                    float v = (wval[j] && ci < Cin)
                                  ? wrptr[j][(size_t)ci * kk2 + kpos] : 0.f;
                    tmp4[j] = f2bf(v);
                }
                __builtin_memcpy(&sha[bufc][4 * tid], tmp4, 8);   // ds_store_b64
                __syncthreads();

                v16bf fa;
                __builtin_memcpy(&fa, &sha[bufc][lane * 16], 32); // 2x ds_load_b128
                bufc ^= 1;

                // --- per-lane activation gathers (coalesced across lanes) ---
                v16bf fb0, fb1;
#pragma unroll
                for (int e = 0; e < 16; ++e) {
                    const int  ci  = cc + koff(e, half);
                    const bool civ = (ci < Cin);
                    fb0[e] = f2bf((v0 && civ) ? xim[(size_t)ci * HWi + off0] : 0.f);
                    fb1[e] = f2bf((v1 && civ) ? xim[(size_t)ci * HWi + off1] : 0.f);
                }
                acc0 = __builtin_amdgcn_wmma_f32_16x16x32_bf16(false, fa, false, fb0,
                                                               (short)0, acc0, false, false);
                acc1 = __builtin_amdgcn_wmma_f32_16x16x32_bf16(false, fa, false, fb1,
                                                               (short)0, acc1, false, false);
            }
        }
    }

#pragma unroll
    for (int r = 0; r < 8; ++r) {
        const int m = m0 + half * 8 + r;
        if (m < Cout) {
            const float bb = bias ? bias[m] : 0.f;
            float* yrow = y + ((size_t)img * Cout + m) * HWo;
            if (sv0) yrow[s0] = acc0[r] + bb;
            if (sv1) yrow[s1] = acc1[r] + bb;
        }
    }
}

// ---------------------------------------------------------------------------
// GEMM: C[M,N] = A[M,K] * B[N,K]^T (+ bias[N]).  bf16 WMMA, f32 accumulate.
// REQUIRES: M % 64 == 0, N % 32 == 0, K % 32 == 0 (true for all SSM GEMMs:
// M=40960, N,K in {192,384}). No masks; fragments loaded as float4 vectors
// (16B-aligned: row base is a multiple of K>=32 floats, kk%32==0, half*8%8==0).
// One wave computes a 16(M) x 32(N) tile = 2 v_wmma per K-chunk.
// The A operand streams once from HBM; the next K-chunk is software-
// prefetched (global_prefetch_b8, ISA 10.5: no VGPR return, no counter) to
// overlap HBM latency with the current chunk's WMMA work.
// ---------------------------------------------------------------------------
__global__ __launch_bounds__(128)
void gemm_wmma_k(const float* __restrict__ A, const float* __restrict__ Bw,
                 const float* __restrict__ bias, float* __restrict__ C,
                 int M, int N, int K)
{
    const int lane = threadIdx.x & 31;
    const int wv   = threadIdx.x >> 5;
    const int l16  = lane & 15;
    const int half = lane >> 4;
    const int m0 = (blockIdx.y * 4 + wv) * 16;
    const int n0 = blockIdx.x * 32;

    const float* arow  = A  + (size_t)(m0 + l16) * K;
    const float* brow0 = Bw + (size_t)(n0 + l16) * K;
    const float* brow1 = Bw + (size_t)(n0 + 16 + l16) * K;

    v8f acc0 = {0.f, 0.f, 0.f, 0.f, 0.f, 0.f, 0.f, 0.f};
    v8f acc1 = {0.f, 0.f, 0.f, 0.f, 0.f, 0.f, 0.f, 0.f};

    for (int kk = 0; kk < K; kk += 32) {
        const int base = kk + half * 8;      // this lane's first K (elems 0..7)
        if (kk + 32 < K) {                   // wave-uniform guard
            __builtin_prefetch(arow + base + 32, 0, 1);   // global_prefetch_b8
        }
        f32x4 a0 = *(const f32x4*)(arow  + base);
        f32x4 a1 = *(const f32x4*)(arow  + base + 4);
        f32x4 a2 = *(const f32x4*)(arow  + base + 16);
        f32x4 a3 = *(const f32x4*)(arow  + base + 20);
        f32x4 b0 = *(const f32x4*)(brow0 + base);
        f32x4 b1 = *(const f32x4*)(brow0 + base + 4);
        f32x4 b2 = *(const f32x4*)(brow0 + base + 16);
        f32x4 b3 = *(const f32x4*)(brow0 + base + 20);
        f32x4 c0 = *(const f32x4*)(brow1 + base);
        f32x4 c1 = *(const f32x4*)(brow1 + base + 4);
        f32x4 c2 = *(const f32x4*)(brow1 + base + 16);
        f32x4 c3 = *(const f32x4*)(brow1 + base + 20);
        v16bf fa, fb0, fb1;
#pragma unroll
        for (int e = 0; e < 4; ++e) {
            fa[e]       = f2bf(a0[e]);  fa[e + 4]   = f2bf(a1[e]);
            fa[e + 8]   = f2bf(a2[e]);  fa[e + 12]  = f2bf(a3[e]);
            fb0[e]      = f2bf(b0[e]);  fb0[e + 4]  = f2bf(b1[e]);
            fb0[e + 8]  = f2bf(b2[e]);  fb0[e + 12] = f2bf(b3[e]);
            fb1[e]      = f2bf(c0[e]);  fb1[e + 4]  = f2bf(c1[e]);
            fb1[e + 8]  = f2bf(c2[e]);  fb1[e + 12] = f2bf(c3[e]);
        }
        acc0 = __builtin_amdgcn_wmma_f32_16x16x32_bf16(false, fa, false, fb0,
                                                       (short)0, acc0, false, false);
        acc1 = __builtin_amdgcn_wmma_f32_16x16x32_bf16(false, fa, false, fb1,
                                                       (short)0, acc1, false, false);
    }

    const int na = n0 + l16;
    const int nb = na + 16;
    const float ba = bias ? bias[na] : 0.f;
    const float bb = bias ? bias[nb] : 0.f;
#pragma unroll
    for (int r = 0; r < 8; ++r) {
        const int m = m0 + half * 8 + r;
        C[(size_t)m * N + na] = acc0[r] + ba;
        C[(size_t)m * N + nb] = acc1[r] + bb;
    }
}

// ---------------------------------------------------------------------------
// GroupNorm stats: one block per (n, group); group data is contiguous.
// stats[2*bid] = mean, stats[2*bid+1] = var (population)
// ---------------------------------------------------------------------------
__global__ __launch_bounds__(256)
void gn_stats_k(const float* __restrict__ x, float* __restrict__ stats,
                int Gr, int cpg, int HW, int C)
{
    const int n = blockIdx.x / Gr;
    const int g = blockIdx.x % Gr;
    const size_t base = ((size_t)n * C + (size_t)g * cpg) * HW;
    const int cnt = cpg * HW;
    float s = 0.f, s2 = 0.f;
    for (int i = threadIdx.x; i < cnt; i += blockDim.x) {
        float v = x[base + i];
        s += v; s2 += v * v;
    }
    __shared__ float sh[256], sh2[256];
    sh[threadIdx.x] = s; sh2[threadIdx.x] = s2;
    __syncthreads();
    for (int o = 128; o > 0; o >>= 1) {
        if ((int)threadIdx.x < o) {
            sh[threadIdx.x]  += sh[threadIdx.x + o];
            sh2[threadIdx.x] += sh2[threadIdx.x + o];
        }
        __syncthreads();
    }
    if (threadIdx.x == 0) {
        float m = sh[0] / cnt;
        float v = sh2[0] / cnt - m * m;
        stats[2 * blockIdx.x + 0] = m;
        stats[2 * blockIdx.x + 1] = v;
    }
}

__global__ __launch_bounds__(256)
void gn_apply_gelu_k(float* __restrict__ x, const float* __restrict__ stats,
                     const float* __restrict__ gamma, const float* __restrict__ beta,
                     int N, int C, int HW, int Gr, int cpg)
{
    int idx = blockIdx.x * blockDim.x + threadIdx.x;
    int tot = N * C * HW;
    if (idx >= tot) return;
    int n = idx / (C * HW);
    int r = idx - n * C * HW;
    int c = r / HW;
    int g = c / cpg;
    float m = stats[2 * (n * Gr + g) + 0];
    float v = stats[2 * (n * Gr + g) + 1];
    float y = (x[idx] - m) * rsqrtf(v + 1e-5f) * gamma[c] + beta[c];
    x[idx] = gelu_exact(y);
}

// ---------------------------------------------------------------------------
// Elementwise helpers
// ---------------------------------------------------------------------------
__global__ __launch_bounds__(256)
void add_k(const float* __restrict__ a, const float* __restrict__ b,
           float* __restrict__ out, int n)
{
    int i = blockIdx.x * blockDim.x + threadIdx.x;
    if (i < n) out[i] = a[i] + b[i];
}

__global__ __launch_bounds__(256)
void gelu_k(float* __restrict__ x, int n)
{
    int i = blockIdx.x * blockDim.x + threadIdx.x;
    if (i < n) x[i] = gelu_exact(x[i]);
}

// cloud head output: 1x1 conv 32->1 + bias + sigmoid
__global__ __launch_bounds__(256)
void cloud_out_k(const float* __restrict__ x, const float* __restrict__ w,
                 const float* __restrict__ b, float* __restrict__ sup,
                 int N, int Cin, int HW)
{
    int idx = blockIdx.x * blockDim.x + threadIdx.x;
    if (idx >= N * HW) return;
    int n = idx / HW;
    int p = idx - n * HW;
    float acc = b[0];
    for (int c = 0; c < Cin; ++c)
        acc += w[c] * x[((size_t)n * Cin + c) * HW + p];
    sup[idx] = sigm(acc);
}

// build SSM input: x[(seq*T+t)*C+c] = f3[b,t,c,hw] * (1 - sup[b,t,hw])
__global__ __launch_bounds__(256)
void build_x_k(const float* __restrict__ f3, const float* __restrict__ sup,
               float* __restrict__ x, int B, int T, int C, int HW)
{
    int idx = blockIdx.x * blockDim.x + threadIdx.x;
    int tot = B * HW * T * C;
    if (idx >= tot) return;
    int row = idx / C;
    int c   = idx - row * C;
    int seq = row / T;
    int t   = row - seq * T;
    int b   = seq / HW;
    int pix = seq - b * HW;
    float s = sup[((size_t)(b * T + t)) * HW + pix];
    x[idx] = f3[(((size_t)(b * T + t) * C + c)) * HW + pix] * (1.0f - s);
}

// gate: g = sigmoid(bg_out) * sigmoid(sg_w[c]*sup + sg_b[c])   (in place on g)
__global__ __launch_bounds__(256)
void gate_k(float* __restrict__ g, const float* __restrict__ sup,
            const float* __restrict__ sgw, const float* __restrict__ sgb,
            int B, int T, int C, int HW)
{
    int idx = blockIdx.x * blockDim.x + threadIdx.x;
    int tot = B * HW * T * C;
    if (idx >= tot) return;
    int row = idx / C;
    int c   = idx - row * C;
    int seq = row / T;
    int t   = row - seq * T;
    int b   = seq / HW;
    int pix = seq - b * HW;
    float s = sup[((size_t)(b * T + t)) * HW + pix];
    g[idx] = sigm(g[idx]) * sigm(sgw[c] * s + sgb[c]);
}

// recurrent scan over T:  s = (1-g)*s + g*h
__global__ __launch_bounds__(256)
void scan_k(const float* __restrict__ h, const float* __restrict__ g,
            float* __restrict__ sseq, int NSEQ, int T, int C)
{
    int idx = blockIdx.x * blockDim.x + threadIdx.x;
    if (idx >= NSEQ * C) return;
    int seq = idx / C;
    int c   = idx - seq * C;
    float s = 0.f;
    for (int t = 0; t < T; ++t) {
        size_t o = ((size_t)(seq * T + t)) * C + c;
        float gv = g[o];
        s = (1.0f - gv) * s + gv * h[o];
        sseq[o] = s;
    }
}

// mean over T, reshape to NCHW: out[b,c,hw] = mean_t x[(b*HW+hw)*T+t, c]
__global__ __launch_bounds__(256)
void mean_t_k(const float* __restrict__ x, float* __restrict__ out,
              int B, int T, int C, int HW)
{
    int idx = blockIdx.x * blockDim.x + threadIdx.x;
    if (idx >= B * C * HW) return;
    int b = idx / (C * HW);
    int r = idx - b * C * HW;
    int c = r / HW;
    int pix = r - c * HW;
    int seq = b * HW + pix;
    float s = 0.f;
    for (int t = 0; t < T; ++t)
        s += x[((size_t)(seq * T + t)) * C + c];
    out[idx] = s * (1.0f / T);
}

// weighted skip fuse over T
__global__ __launch_bounds__(256)
void wsf_k(const float* __restrict__ feats, const float* __restrict__ sup,
           float* __restrict__ out, int B, int T, int C, int HW)
{
    int idx = blockIdx.x * blockDim.x + threadIdx.x;
    if (idx >= B * C * HW) return;
    int b = idx / (C * HW);
    int r = idx - b * C * HW;
    int c = r / HW;
    int pix = r - c * HW;
    float csum = 0.f;
    for (int t = 0; t < T; ++t) {
        float cf = 1.0f - sup[((size_t)(b * T + t)) * HW + pix];
        cf = fminf(fmaxf(cf, 0.0f), 1.0f);
        csum += cf;
    }
    float inv = 1.0f / (csum + 1e-6f);
    float acc = 0.f;
    for (int t = 0; t < T; ++t) {
        float cf = 1.0f - sup[((size_t)(b * T + t)) * HW + pix];
        cf = fminf(fmaxf(cf, 0.0f), 1.0f);
        acc += feats[(((size_t)(b * T + t) * C + c)) * HW + pix] * (cf * inv);
    }
    out[idx] = acc;
}

// bilinear resize, half-pixel centers, edge clamp (align_corners=False)
__global__ __launch_bounds__(256)
void resize_k(const float* __restrict__ x, float* __restrict__ y,
              int N, int C, int Hin, int Win, int Hout, int Wout)
{
    int idx = blockIdx.x * blockDim.x + threadIdx.x;
    int tot = N * C * Hout * Wout;
    if (idx >= tot) return;
    int nc = idx / (Hout * Wout);
    int r  = idx - nc * (Hout * Wout);
    int oy = r / Wout;
    int ox = r - oy * Wout;
    float sy = (oy + 0.5f) * ((float)Hin / Hout) - 0.5f;
    float sx = (ox + 0.5f) * ((float)Win / Wout) - 0.5f;
    int y0 = (int)floorf(sy), x0 = (int)floorf(sx);
    float ty = sy - y0, tx = sx - x0;
    int y0c = y0 < 0 ? 0 : (y0 > Hin - 1 ? Hin - 1 : y0);
    int y1c = (y0 + 1) < 0 ? 0 : ((y0 + 1) > Hin - 1 ? Hin - 1 : y0 + 1);
    int x0c = x0 < 0 ? 0 : (x0 > Win - 1 ? Win - 1 : x0);
    int x1c = (x0 + 1) < 0 ? 0 : ((x0 + 1) > Win - 1 ? Win - 1 : x0 + 1);
    const float* p = x + (size_t)nc * Hin * Win;
    float v00 = p[(size_t)y0c * Win + x0c], v01 = p[(size_t)y0c * Win + x1c];
    float v10 = p[(size_t)y1c * Win + x0c], v11 = p[(size_t)y1c * Win + x1c];
    float v0 = v00 + (v01 - v00) * tx;
    float v1 = v10 + (v11 - v10) * tx;
    y[idx] = v0 + (v1 - v0) * ty;
}

// channel concat: out[n, 0:C1]=a, out[n, C1:C1+C2]=b
__global__ __launch_bounds__(256)
void concat_k(const float* __restrict__ a, const float* __restrict__ b,
              float* __restrict__ out, int N, int C1, int C2, int HW)
{
    int idx = blockIdx.x * blockDim.x + threadIdx.x;
    int Ct = C1 + C2;
    int tot = N * Ct * HW;
    if (idx >= tot) return;
    int n = idx / (Ct * HW);
    int r = idx - n * Ct * HW;
    int c = r / HW;
    int pix = r - c * HW;
    float v;
    if (c < C1) v = a[((size_t)n * C1 + c) * HW + pix];
    else        v = b[((size_t)n * C2 + (c - C1)) * HW + pix];
    out[idx] = v;
}

// ---------------------------------------------------------------------------
// Host orchestration
// ---------------------------------------------------------------------------
struct Pg { const float *w, *g, *b; };
struct Rp { Pg c1, c2; };
struct Lp { const float *w, *b; };
struct Blk { const float *in_w, *out_w; Lp bg, sg, f1, f2; };

static int ngroups(int c) {
    int g = c / 8;
    if (g < 1) g = 1;
    if (g > 32) g = 32;
    return g;
}

extern "C" void kernel_launch(void* const* d_in, const int* in_sizes, int n_in,
                              void* d_out, int out_size, void* d_ws, size_t ws_size,
                              hipStream_t stream)
{
    (void)in_sizes; (void)n_in; (void)out_size; (void)ws_size;

    // ---- parse inputs in setup_inputs() dict insertion order ----
    int ii = 0;
    auto nxt = [&]() { return (const float*)d_in[ii++]; };
    auto gp  = [&]() { Pg p; p.w = nxt(); p.g = nxt(); p.b = nxt(); return p; };
    auto rp  = [&]() { Rp r; r.c1 = gp(); r.c2 = gp(); return r; };
    auto lp  = [&]() { Lp l; l.w = nxt(); l.b = nxt(); return l; };

    const float* input = nxt();                               // [2,5,3,256,256]
    Pg stem_c = gp(); Rp stem_r = rp();
    Pg d1_c = gp();   Rp d1_r = rp();
    Pg d2_c = gp();   Rp d2_r = rp();
    Pg deep_c = gp(); Rp deep_r = rp();
    Pg c0 = gp(), cc1 = gp(), cc2 = gp();
    const float* cow = nxt(); const float* cob = nxt();
    Blk blk[2];
    for (int i = 0; i < 2; ++i) {
        blk[i].in_w = nxt(); blk[i].out_w = nxt();
        blk[i].bg = lp(); blk[i].sg = lp(); blk[i].f1 = lp(); blk[i].f2 = lp();
    }
    Pg post_c = gp(); Rp post_r = rp();
    Pg m4_c = gp(); Rp m4_r = rp();
    Pg u2_c = gp(); Rp u2_r = rp();
    Pg m2_c = gp(); Rp m2_r = rp();
    Pg u1_c = gp(); Rp u1_r = rp();
    Pg oh_c = gp();
    const float* dow = nxt(); const float* dob = nxt();

    // ---- workspace arena ----
    size_t cur = 0;
    auto carve = [&](size_t elems) {
        float* p = (float*)((char*)d_ws + cur);
        cur += ((elems * sizeof(float) + 255) & ~(size_t)255);
        return p;
    };
    const size_t Q = 7864320;               // 10*48*128*128 == 8192*5*192
    float* A0 = carve(31457280);            // 10*48*256*256
    float* A1 = carve(31457280);
    float* A2 = carve(31457280);
    float* F1 = carve(7864320);             // [10,48,128,128]
    float* F2 = carve(3932160);             // [10,96,64,64]
    float* F3 = carve(7864320);             // [10,192,64,64]
    float* SUP   = carve(40960);            // [10,1,64,64]
    float* SUPUP = carve(163840);           // [10,1,128,128]
    float* DEEP  = carve(1572864);          // [2,192,64,64]
    float* S2    = carve(786432);           // [2,96,64,64]
    float* S1    = carve(1572864);          // [2,48,128,128]
    float* STATS = carve(1024);

    // ---- host-side launch helpers ----
    auto conv = [&](const float* x, const float* w, const float* bias, float* y,
                    int N, int Cin, int Hin, int Win, int Cout, int stride, int k) {
        int pad = k / 2;
        int Hout = (Hin + 2 * pad - k) / stride + 1;
        int Wout = (Win + 2 * pad - k) / stride + 1;
        dim3 grid(CEILDIV(Hout * Wout, 128), CEILDIV(Cout, 16), N);
        conv_wmma_k<<<grid, 128, 0, stream>>>(x, w, bias, y, N, Cin, Hin, Win,
                                              Cout, Hout, Wout, stride, pad, k);
    };
    auto gna = [&](const float* x, const Pg& p, float* y,
                   int N, int Cin, int Hin, int Win, int Cout, int stride) {
        conv(x, p.w, nullptr, y, N, Cin, Hin, Win, Cout, stride, 3);
        int Hout = (Hin + 2 - 3) / stride + 1;
        int Wout = (Win + 2 - 3) / stride + 1;
        int HW = Hout * Wout;
        int Gr = ngroups(Cout), cpg = Cout / Gr;
        gn_stats_k<<<N * Gr, 256, 0, stream>>>(y, STATS, Gr, cpg, HW, Cout);
        int tot = N * Cout * HW;
        gn_apply_gelu_k<<<CEILDIV(tot, 256), 256, 0, stream>>>(y, STATS, p.g, p.b,
                                                               N, Cout, HW, Gr, cpg);
    };
    auto addv = [&](const float* a, const float* b, float* o, int n) {
        add_k<<<CEILDIV(n, 256), 256, 0, stream>>>(a, b, o, n);
    };
    auto resblock = [&](const float* x, const Rp& r, float* t1, float* t2, float* out,
                        int N, int C, int H, int W) {
        gna(x,  r.c1, t1, N, C, H, W, C, 1);
        gna(t1, r.c2, t2, N, C, H, W, C, 1);
        addv(x, t2, out, N * C * H * W);
    };
    auto gemm = [&](const float* A, const float* Bw, const float* bias, float* C,
                    int M, int N, int K) {
        dim3 grid(CEILDIV(N, 32), CEILDIV(M, 64));
        gemm_wmma_k<<<grid, 128, 0, stream>>>(A, Bw, bias, C, M, N, K);
    };

    // =========================== encoder (N=10) ===========================
    const int BT = 10;
    gna(input, stem_c, A0, BT, 3, 256, 256, 48, 1);                 // A0 [10,48,256,256]
    resblock(A0, stem_r, A1, A2, A0, BT, 48, 256, 256);             // x0 = A0
    gna(A0, d1_c, A1, BT, 48, 256, 256, 48, 2);                     // A1 [10,48,128,128]
    resblock(A1, d1_r, A2, A2 + Q, F1, BT, 48, 128, 128);           // f1 = F1
    gna(F1, d2_c, A0, BT, 48, 128, 128, 96, 2);                     // A0 [10,96,64,64]
    resblock(A0, d2_r, A1, A1 + 3932160, F2, BT, 96, 64, 64);       // f2 = F2
    gna(F2, deep_c, A0, BT, 96, 64, 64, 192, 1);                    // A0 [10,192,64,64]
    resblock(A0, deep_r, A1, A1 + Q, F3, BT, 192, 64, 64);          // f3 = F3

    // =========================== cloud head ===============================
    gna(input, c0, A0, BT, 3, 256, 256, 32, 1);
    gna(A0, cc1, A1, BT, 32, 256, 256, 32, 2);
    gna(A1, cc2, A2, BT, 32, 128, 128, 32, 2);                      // [10,32,64,64]
    cloud_out_k<<<CEILDIV(BT * 4096, 256), 256, 0, stream>>>(A2, cow, cob, SUP,
                                                             BT, 32, 4096);

    // =========================== fuse_deep (SSM) ==========================
    const int B = 2, T = 5, C = 192, HW = 4096, NSEQ = B * HW;      // 8192
    const int MT = NSEQ * T;                                        // 40960
    float* X    = A0;
    float* XIN  = A0 + Q;
    float* Hb   = A0 + 2 * Q;
    float* Gb   = A0 + 3 * Q;
    float* HID  = A1;                                               // [40960,384]
    float* SSEQ = A1 + 2 * Q;
    build_x_k<<<CEILDIV(MT * C, 256), 256, 0, stream>>>(F3, SUP, X, B, T, C, HW);
    for (int bi = 0; bi < 2; ++bi) {
        gemm(X, blk[bi].in_w, nullptr, XIN, MT, C, C);              // xin
        gemm(XIN, blk[bi].f1.w, blk[bi].f1.b, HID, MT, 384, C);     // f1
        gelu_k<<<CEILDIV(MT * 384, 256), 256, 0, stream>>>(HID, MT * 384);
        gemm(HID, blk[bi].f2.w, blk[bi].f2.b, Hb, MT, C, 384);      // f2
        addv(XIN, Hb, Hb, MT * C);                                  // h = xin + ffn
        gemm(Hb, blk[bi].bg.w, blk[bi].bg.b, Gb, MT, C, C);         // bg(h)
        gate_k<<<CEILDIV(MT * C, 256), 256, 0, stream>>>(Gb, SUP, blk[bi].sg.w,
                                                         blk[bi].sg.b, B, T, C, HW);
        scan_k<<<CEILDIV(NSEQ * C, 256), 256, 0, stream>>>(Hb, Gb, SSEQ, NSEQ, T, C);
        gemm(SSEQ, blk[bi].out_w, nullptr, X, MT, C, C);            // y -> next x
    }
    float* DEEPB = A2;                       // [2,192,64,64]
    float* PT1   = A2 + 1 * 1572864;
    float* PT2   = A2 + 2 * 1572864;
    float* PT3   = A2 + 3 * 1572864;
    mean_t_k<<<CEILDIV(B * C * HW, 256), 256, 0, stream>>>(X, DEEPB, B, T, C, HW);
    gna(DEEPB, post_c, PT1, B, 192, 64, 64, 192, 1);
    resblock(PT1, post_r, PT2, PT3, DEEP, B, 192, 64, 64);          // deep = DEEP

    // =========================== skip fusion ==============================
    wsf_k<<<CEILDIV(B * 96 * 4096, 256), 256, 0, stream>>>(F2, SUP, S2, B, T, 96, 4096);
    resize_k<<<CEILDIV(BT * 16384, 256), 256, 0, stream>>>(SUP, SUPUP, BT, 1,
                                                           64, 64, 128, 128);
    wsf_k<<<CEILDIV(B * 48 * 16384, 256), 256, 0, stream>>>(F1, SUPUP, S1, B, T,
                                                            48, 16384);

    // =========================== decoder ==================================
    concat_k<<<CEILDIV(B * 288 * 4096, 256), 256, 0, stream>>>(DEEP, S2, A0,
                                                               B, 192, 96, 4096);
    gna(A0, m4_c, A1, B, 288, 64, 64, 192, 1);
    resblock(A1, m4_r, A2, A2 + 1572864, A1, B, 192, 64, 64);
    resize_k<<<CEILDIV(B * 192 * 16384, 256), 256, 0, stream>>>(A1, A0, B, 192,
                                                                64, 64, 128, 128);
    gna(A0, u2_c, A1, B, 192, 128, 128, 96, 1);
    resblock(A1, u2_r, A2, A2 + 3145728, A1, B, 96, 128, 128);
    concat_k<<<CEILDIV(B * 144 * 16384, 256), 256, 0, stream>>>(A1, S1, A0,
                                                                B, 96, 48, 16384);
    gna(A0, m2_c, A1, B, 144, 128, 128, 96, 1);
    resblock(A1, m2_r, A2, A2 + 3145728, A1, B, 96, 128, 128);
    resize_k<<<CEILDIV(B * 96 * 65536, 256), 256, 0, stream>>>(A1, A0, B, 96,
                                                               128, 128, 256, 256);
    gna(A0, u1_c, A1, B, 96, 256, 256, 48, 1);
    resblock(A1, u1_r, A2, A2 + 6291456, A1, B, 48, 256, 256);
    gna(A1, oh_c, A0, B, 48, 256, 256, 48, 1);
    conv(A0, dow, dob, (float*)d_out, B, 48, 256, 256, 3, 1, 3);    // [2,3,256,256]
}
